// T2SPrefillWrapper_25735444037685
// MI455X (gfx1250) — compile-verified, matrix operand order checked by reading, not run on hardware
//
#include <hip/hip_runtime.h>
#include <hip/hip_bf16.h>

// ---------------------------------------------------------------------------
// Model constants (from reference)
// ---------------------------------------------------------------------------
#define DMODEL   512
#define NHEAD    16
#define HD       32
#define NLAYER   24
#define FFDIM    2048
#define TOTAL    1024
#define XCAP     512
#define PCAP     512
#define MAXDEC   1500
#define CACHE_T  (TOTAL + MAXDEC)        // 2524
#define VOCAB    1025
#define BERTD    1024
#define REFCAP   256

typedef _Float16 f16_t;
typedef __attribute__((ext_vector_type(4)))  _Float16 v4h;
typedef __attribute__((ext_vector_type(8)))  _Float16 v8h;
typedef __attribute__((ext_vector_type(16))) _Float16 v16h;
typedef __attribute__((ext_vector_type(8)))  float    v8f;

// ---------------------------------------------------------------------------
// WMMA fragment helpers (CDNA5 ISA 7.12.2, 16-bit A 16x32 / B 32x16):
//   lane L holds row (A) / col (B^T) = L%16,
//   halves j=0..7  -> K = 8*(L>=16)+j ; j=8..15 -> K = 16+8*(L>=16)+(j-8)
// => two contiguous 16-byte LDS chunks at {base, base+32B}.
// C/D f32 16x16: lane L, slot r -> M = r + 8*(L>=16), N = L%16.
// ---------------------------------------------------------------------------
__device__ __forceinline__ v16h load_frag(const f16_t* __restrict__ base,
                                          int row0, int stride) {
  int lane = threadIdx.x & 31;
  const f16_t* p = base + (row0 + (lane & 15)) * stride + ((lane >> 4) << 3);
  v8h lo = *(const v8h*)p;
  v8h hi = *(const v8h*)(p + 16);
  return __builtin_shufflevector(lo, hi, 0,1,2,3,4,5,6,7,8,9,10,11,12,13,14,15);
}

__device__ __forceinline__ v8f wmma16(v16h a, v16h b, v8f c) {
  return __builtin_amdgcn_wmma_f32_16x16x32_f16(false, a, false, b,
                                                (short)0, c, false, false);
}

// ---------------------------------------------------------------------------
// Async global->LDS staging of a 128-row x 64-half (128B/row) f16 tile.
// Per wave: 4 x GLOBAL_LOAD_ASYNC_TO_LDS_B128 (each lane moves 16B).
// LDS row pitch = 72 halves (144B) -> conflict-free ds_load_b128 frag reads.
// ---------------------------------------------------------------------------
#define TPITCH 72
__device__ __forceinline__ void stage_async(const f16_t* __restrict__ g,
                                            int row0, int k0, int Kld,
                                            unsigned ldsbase, int wave, int lane) {
  unsigned long long base = (unsigned long long)(const void*)g;
#pragma unroll
  for (int i = 0; i < 4; ++i) {
    int c = (wave * 4 + i) * 32 + lane;        // 0..1023 16B-chunk id
    int r = c >> 3, kk = c & 7;                // row 0..127, 16B chunk 0..7
    unsigned goff = ((unsigned)(row0 + r) * (unsigned)Kld +
                     (unsigned)k0 + (unsigned)(kk * 8)) * 2u;
    unsigned laddr = ldsbase + (unsigned)(r * (TPITCH * 2) + kk * 16);
    asm volatile("global_load_async_to_lds_b128 %0, %1, %2"
                 :: "v"(laddr), "v"(goff), "s"(base) : "memory");
  }
}

__device__ __forceinline__ void wait_async0() {
  asm volatile("s_wait_asynccnt 0x0" ::: "memory");
}

// ---------------------------------------------------------------------------
// Double-buffered async WMMA GEMM: C[M,N] = A[M,K] @ B[K,N] (+bias)(+res)(relu)
// A: f16 row-major [M][K]; Bt: f16 row-major [N][K] (pre-transposed weights).
// Block = 256 threads (8 waves); tile 128x128, K stage 64 (2 sub-steps of 32).
// Wave (wm,wn) grid 2x4: 64 rows x 32 cols => 8 accumulators, 16 WMMA/stage.
// M,N multiples of 128; K multiple of 64 (true for all uses here).
// ---------------------------------------------------------------------------
template<bool BIAS, bool RELU, bool RES>
__global__ __launch_bounds__(256)
void gemm_async(const f16_t* __restrict__ A, const f16_t* __restrict__ Bt,
                const float* __restrict__ bias, const float* __restrict__ res,
                float* __restrict__ C, int M, int N, int K) {
  __shared__ __align__(16) f16_t As[2][128 * TPITCH];
  __shared__ __align__(16) f16_t Bs[2][128 * TPITCH];
  const int n0 = blockIdx.x * 128;
  const int m0 = blockIdx.y * 128;
  const int tid = threadIdx.x, wave = tid >> 5, lane = tid & 31;
  const int wm = (wave & 1) * 64;
  const int wn = (wave >> 1) * 32;

  v8f acc[4][2];
#pragma unroll
  for (int i = 0; i < 4; ++i)
#pragma unroll
    for (int j = 0; j < 2; ++j) acc[i][j] = (v8f)(0.f);

  // prologue: stage 0 into buffer 0
  stage_async(A,  m0, 0, K, (unsigned)(size_t)(void*)&As[0][0], wave, lane);
  stage_async(Bt, n0, 0, K, (unsigned)(size_t)(void*)&Bs[0][0], wave, lane);

  const int nst = K >> 6;
  for (int st = 0; st < nst; ++st) {
    wait_async0();
    __syncthreads();
    if (st + 1 < nst) {                        // stream next stage
      int nb = (st + 1) & 1;
      stage_async(A,  m0, (st + 1) * 64, K,
                  (unsigned)(size_t)(void*)&As[nb][0], wave, lane);
      stage_async(Bt, n0, (st + 1) * 64, K,
                  (unsigned)(size_t)(void*)&Bs[nb][0], wave, lane);
    }
    const f16_t* as = &As[st & 1][0];
    const f16_t* bs = &Bs[st & 1][0];
#pragma unroll
    for (int ks = 0; ks < 2; ++ks) {
      const f16_t* ap = as + ks * 32;
      const f16_t* bp = bs + ks * 32;
      v16h af[4], bf[2];
#pragma unroll
      for (int fi = 0; fi < 4; ++fi) af[fi] = load_frag(ap, wm + fi * 16, TPITCH);
#pragma unroll
      for (int fj = 0; fj < 2; ++fj) bf[fj] = load_frag(bp, wn + fj * 16, TPITCH);
#pragma unroll
      for (int fi = 0; fi < 4; ++fi)
#pragma unroll
        for (int fj = 0; fj < 2; ++fj)
          acc[fi][fj] = wmma16(af[fi], bf[fj], acc[fi][fj]);
    }
  }

  const int nloc = lane & 15;
  const int mb = (lane >> 4) << 3;
#pragma unroll
  for (int fi = 0; fi < 4; ++fi)
#pragma unroll
    for (int fj = 0; fj < 2; ++fj) {
      int gn = n0 + wn + fj * 16 + nloc;
      float bv = BIAS ? bias[gn] : 0.f;
#pragma unroll
      for (int r = 0; r < 8; ++r) {
        int gm = m0 + wm + fi * 16 + mb + r;
        float v = acc[fi][fj][r] + bv;
        if (RES)  v += res[(size_t)gm * N + gn];
        if (RELU) v = v > 0.f ? v : 0.f;
        C[(size_t)gm * N + gn] = v;
      }
    }
}

// ---------------------------------------------------------------------------
// f32 -> f16 converters (n multiple of 4)
// ---------------------------------------------------------------------------
__global__ __launch_bounds__(256)
void cvt_kernel(const float* __restrict__ in, f16_t* __restrict__ out, int n) {
  int i = (blockIdx.x * 256 + threadIdx.x) * 4;
  if (i >= n) return;
  float4 v = *(const float4*)(in + i);
  v4h o; o[0] = (f16_t)v.x; o[1] = (f16_t)v.y; o[2] = (f16_t)v.z; o[3] = (f16_t)v.w;
  *(v4h*)(out + i) = o;
}

// out[n*K+k] = (f16)in[k*N+n]; K,N multiples of 64. LDS-tiled transpose.
__global__ __launch_bounds__(256)
void cvtT_kernel(const float* __restrict__ in, f16_t* __restrict__ out,
                 int K, int N) {
  __shared__ f16_t t[64][65];
  const int nb = blockIdx.x * 64, kb = blockIdx.y * 64;
  const int tid = threadIdx.x;
#pragma unroll
  for (int i = 0; i < 16; ++i) {
    int e = tid + i * 256;
    int kk = e >> 6, nn = e & 63;
    t[kk][nn] = (f16_t)in[(size_t)(kb + kk) * N + nb + nn];
  }
  __syncthreads();
#pragma unroll
  for (int i = 0; i < 16; ++i) {
    int e = tid + i * 256;
    int nn = e >> 6, kk = e & 63;
    out[(size_t)(nb + nn) * K + kb + kk] = t[kk][nn];
  }
}

// ---------------------------------------------------------------------------
// Attention mask (matches reference "invalid")
// ---------------------------------------------------------------------------
__device__ __forceinline__ bool mask_invalid(int q, int k, int xl, int tot) {
  if (q >= tot || k >= tot) return true;
  if (k < xl) return false;     // everyone sees the text region
  if (q < xl) return true;      // text queries cannot see the prompt region
  return k > q;                 // causal within the prompt region
}

// ---------------------------------------------------------------------------
// Flash-style attention (one head / 64 query rows per block; 4 waves).
// qkv layout: [TOTAL][3*DMODEL] = [q | k | v]; out: [TOTAL][DMODEL]
// ---------------------------------------------------------------------------
__global__ __launch_bounds__(128)
void attn_kernel(const float* __restrict__ qkv, float* __restrict__ outp,
                 const int* __restrict__ rl_, const int* __restrict__ tl_,
                 const int* __restrict__ pl_) {
  __shared__ __align__(16) f16_t Ks[32 * 40];   // [key][hd]
  __shared__ __align__(16) f16_t Vt[32 * 40];   // [hd][key]
  __shared__ __align__(16) f16_t Pw[4][16 * 40];
  const int head = blockIdx.x;
  const int qb   = blockIdx.y;
  const int tid = threadIdx.x, wave = tid >> 5, lane = tid & 31;
  const int q0 = qb * 64 + wave * 16;
  const int xl  = rl_[0] + tl_[0];
  const int tot = xl + pl_[0];
  const float scale = 0.17677669529663687f;     // 1/sqrt(32)

  const int mq = lane & 15;
  const int kb = (lane >> 4) << 3;
  v16h qa;
  {
    const float* qp = qkv + (size_t)(q0 + mq) * (3 * DMODEL) + head * HD;
#pragma unroll
    for (int j = 0; j < 8; ++j) qa[j]     = (f16_t)qp[kb + j];
#pragma unroll
    for (int j = 0; j < 8; ++j) qa[8 + j] = (f16_t)qp[16 + kb + j];
  }

  v8f o1 = (v8f)(0.f), o2 = (v8f)(0.f);
  float mrow[8], lrow[8];
#pragma unroll
  for (int r = 0; r < 8; ++r) { mrow[r] = -3.0e38f; lrow[r] = 0.f; }

  const int ncol = lane & 15;
  const int mb8 = (lane >> 4) << 3;

  for (int kblk = 0; kblk < TOTAL; kblk += 32) {
    __syncthreads();
#pragma unroll
    for (int i = 0; i < 8; ++i) {           // stage K and V^T (32x32 each)
      int e = tid + i * 128;
      int key = e >> 5, hd = e & 31;
      const float* row = qkv + (size_t)(kblk + key) * (3 * DMODEL) + head * HD;
      Ks[key * 40 + hd] = (f16_t)row[DMODEL + hd];
      Vt[hd * 40 + key] = (f16_t)row[2 * DMODEL + hd];
    }
    __syncthreads();

    v8f s[2];
#pragma unroll
    for (int jt = 0; jt < 2; ++jt) {
      v16h bk = load_frag(Ks, jt * 16, 40);
      s[jt] = wmma16(qa, bk, (v8f)(0.f));
    }

#pragma unroll
    for (int r = 0; r < 8; ++r) {
      int qrow = q0 + mb8 + r;
      float best = -3.0e38f;
#pragma unroll
      for (int jt = 0; jt < 2; ++jt) {
        int kcol = kblk + jt * 16 + ncol;
        float v = mask_invalid(qrow, kcol, xl, tot) ? -1.0e9f : s[jt][r] * scale;
        s[jt][r] = v;
        best = fmaxf(best, v);
      }
      for (int sh = 1; sh < 16; sh <<= 1)
        best = fmaxf(best, __shfl_xor(best, sh, 32));
      float mnew = fmaxf(mrow[r], best);
      float alpha = __expf(mrow[r] - mnew);
      mrow[r] = mnew;
      o1[r] *= alpha; o2[r] *= alpha;
      float rsum = 0.f;
#pragma unroll
      for (int jt = 0; jt < 2; ++jt) {
        float p = __expf(s[jt][r] - mnew);
        s[jt][r] = p;
        rsum += p;
      }
      for (int sh = 1; sh < 16; sh <<= 1) rsum += __shfl_xor(rsum, sh, 32);
      lrow[r] = lrow[r] * alpha + rsum;
    }

    f16_t* pw = &Pw[wave][0];
#pragma unroll
    for (int jt = 0; jt < 2; ++jt)
#pragma unroll
      for (int r = 0; r < 8; ++r)
        pw[(mb8 + r) * 40 + jt * 16 + ncol] = (f16_t)s[jt][r];
    // same-wave LDS ops are in-order on CDNA5; safe to reload immediately
    v16h pa  = load_frag(pw, 0, 40);
    v16h bv1 = load_frag(Vt, 0, 40);
    v16h bv2 = load_frag(Vt, 16, 40);
    o1 = wmma16(pa, bv1, o1);
    o2 = wmma16(pa, bv2, o2);
  }

#pragma unroll
  for (int r = 0; r < 8; ++r) {
    int row = q0 + mb8 + r;
    float inv = 1.f / lrow[r];
    outp[(size_t)row * DMODEL + head * HD + ncol]      = o1[r] * inv;
    outp[(size_t)row * DMODEL + head * HD + 16 + ncol] = o2[r] * inv;
  }
}

// ---------------------------------------------------------------------------
// Row LayerNorm over D=512 (biased variance, eps=1e-5)
// ---------------------------------------------------------------------------
__global__ __launch_bounds__(256)
void layernorm_kernel(const float* __restrict__ x, const float* __restrict__ w,
                      const float* __restrict__ b, float* __restrict__ out) {
  const int row = blockIdx.x, tid = threadIdx.x;
  float v0 = x[(size_t)row * DMODEL + tid];
  float v1 = x[(size_t)row * DMODEL + 256 + tid];
  float s = v0 + v1, ss = v0 * v0 + v1 * v1;
  for (int sh = 1; sh < 32; sh <<= 1) {
    s  += __shfl_xor(s, sh, 32);
    ss += __shfl_xor(ss, sh, 32);
  }
  __shared__ float sb[8], ssb[8];
  if ((tid & 31) == 0) { sb[tid >> 5] = s; ssb[tid >> 5] = ss; }
  __syncthreads();
  float ts = 0.f, tss = 0.f;
#pragma unroll
  for (int i = 0; i < 8; ++i) { ts += sb[i]; tss += ssb[i]; }
  float mean = ts * (1.f / DMODEL);
  float var  = tss * (1.f / DMODEL) - mean * mean;
  float rs = rsqrtf(var + 1e-5f);
  out[(size_t)row * DMODEL + tid]       = (v0 - mean) * rs * w[tid] + b[tid];
  out[(size_t)row * DMODEL + 256 + tid] = (v1 - mean) * rs * w[256 + tid] + b[256 + tid];
}

// ---------------------------------------------------------------------------
// Copy k,v slices of qkv into this layer's cache rows [0, TOTAL)
// ---------------------------------------------------------------------------
__global__ __launch_bounds__(256)
void copy_kv_kernel(const float* __restrict__ qkv,
                    float* __restrict__ kc, float* __restrict__ vc) {
  int idx = blockIdx.x * 256 + threadIdx.x;
  if (idx >= TOTAL * DMODEL) return;
  int row = idx >> 9, c = idx & 511;
  const float* src = qkv + (size_t)row * (3 * DMODEL);
  kc[(size_t)row * DMODEL + c] = src[DMODEL + c];
  vc[(size_t)row * DMODEL + c] = src[2 * DMODEL + c];
}

// Zero the MAX_DECODE pad region of both caches (all layers)
__global__ __launch_bounds__(256)
void zero_pad_kernel(float* __restrict__ kc, float* __restrict__ vc) {
  long long i = (long long)blockIdx.x * 256 + threadIdx.x;
  const long long per = (long long)NLAYER * MAXDEC * DMODEL;
  if (i >= per) return;
  int l = (int)(i / ((long long)MAXDEC * DMODEL));
  long long off = i - (long long)l * MAXDEC * DMODEL;
  long long dst = (long long)l * CACHE_T * DMODEL + (long long)TOTAL * DMODEL + off;
  kc[dst] = 0.f;
  vc[dst] = 0.f;
}

// ---------------------------------------------------------------------------
// Build A_bert[512][1024]: row j = ref_bert[:,j] (j<rl), text_bert[:,j-rl]
// (rl<=j<rl+tl), else 0. (ref/text_bert are [1024][256] row-major)
// ---------------------------------------------------------------------------
__global__ __launch_bounds__(256)
void bertA_kernel(const float* __restrict__ ref_bert,
                  const float* __restrict__ text_bert,
                  const int* __restrict__ rl_, const int* __restrict__ tl_,
                  float* __restrict__ A) {
  int idx = blockIdx.x * 256 + threadIdx.x;
  if (idx >= XCAP * BERTD) return;
  int j = idx >> 10, k = idx & 1023;
  int rl = rl_[0], xl = rl + tl_[0];
  float v = 0.f;
  if (j < rl)      v = ref_bert[(size_t)k * REFCAP + j];
  else if (j < xl) v = text_bert[(size_t)k * REFCAP + (j - rl)];
  A[(size_t)j * BERTD + k] = v;
}

// ---------------------------------------------------------------------------
// Embedding + packing + sinusoidal PE  -> h[TOTAL][DMODEL]
// ---------------------------------------------------------------------------
__global__ __launch_bounds__(256)
void embed_kernel(const int* __restrict__ prompts, const int* __restrict__ pl_,
                  const int* __restrict__ ref_seq, const int* __restrict__ rl_,
                  const int* __restrict__ text_seq, const int* __restrict__ tl_,
                  const float* __restrict__ text_emb,
                  const float* __restrict__ audio_emb,
                  const float* __restrict__ bproj, const float* __restrict__ bpb,
                  const float* __restrict__ alpha_t, const float* __restrict__ alpha_a,
                  float* __restrict__ h) {
  const int t = blockIdx.x, tid = threadIdx.x;
  const int rl = rl_[0], tl = tl_[0], pl = pl_[0];
  const int xl = rl + tl, tot = xl + pl;
#pragma unroll
  for (int half = 0; half < 2; ++half) {
    int c = tid + half * 256;
    float val = 0.f;
    if (t < xl) {
      int idx = (t < rl) ? ref_seq[t] : text_seq[t - rl];
      int i = c >> 1;
      float freq = __expf(-(float)(2 * i) * (9.210340371976184f / (float)DMODEL));
      float pe = (c & 1) ? __cosf((float)t * freq) : __sinf((float)t * freq);
      val = text_emb[(size_t)idx * DMODEL + c] + bproj[(size_t)t * DMODEL + c] +
            bpb[c] + alpha_t[0] * pe;
    } else if (t < tot) {
      int j = t - xl;
      int idx = prompts[j];
      int i = c >> 1;
      float freq = __expf(-(float)(2 * i) * (9.210340371976184f / (float)DMODEL));
      float pe = (c & 1) ? __cosf((float)j * freq) : __sinf((float)j * freq);
      val = audio_emb[(size_t)idx * DMODEL + c] + alpha_a[0] * pe;
    }
    h[(size_t)t * DMODEL + c] = val;
  }
}

// ---------------------------------------------------------------------------
// logits[n] = h[tot-1] . predict_w[:,n] + predict_b[n]   (n < 1025)
// ---------------------------------------------------------------------------
__global__ __launch_bounds__(256)
void predict_kernel(const float* __restrict__ h, const float* __restrict__ pw,
                    const float* __restrict__ pb,
                    const int* __restrict__ rl_, const int* __restrict__ tl_,
                    const int* __restrict__ pl_, float* __restrict__ logits) {
  int n = blockIdx.x * 256 + threadIdx.x;
  if (n >= VOCAB) return;
  int tot = rl_[0] + tl_[0] + pl_[0];
  const float* hr = h + (size_t)(tot - 1) * DMODEL;
  float acc = pb[n];
  for (int k = 0; k < DMODEL; ++k) acc += hr[k] * pw[(size_t)k * VOCAB + n];
  logits[n] = acc;
}

__global__ __launch_bounds__(256)
void finalize_kernel(const float* __restrict__ logits,
                     const int* __restrict__ rl_, const int* __restrict__ tl_,
                     const int* __restrict__ pl_, float* __restrict__ out) {
  __shared__ float bv[256];
  __shared__ int   bi[256];
  int tid = threadIdx.x;
  float best = -3.0e38f; int bidx = 0;
  for (int i = tid; i < VOCAB; i += 256) {
    float v = logits[i];
    if (v > best) { best = v; bidx = i; }
  }
  bv[tid] = best; bi[tid] = bidx;
  __syncthreads();
  if (tid == 0) {
    float B = -3.0e38f; int I = VOCAB;
    for (int i = 0; i < 256; ++i) {
      if (bv[i] > B || (bv[i] == B && bi[i] < I)) { B = bv[i]; I = bi[i]; }
    }
    int tot = rl_[0] + tl_[0] + pl_[0];
    out[VOCAB]     = (float)I;                       // sampled
    out[VOCAB + 1] = (I == 1024) ? 1.f : 0.f;        // eos
    out[VOCAB + 2] = (float)tot;                     // active
    out[VOCAB + 3] = (float)pl_[0];                  // next_position
  }
}

// ---------------------------------------------------------------------------
// Host driver
// ---------------------------------------------------------------------------
extern "C" void kernel_launch(void* const* d_in, const int* in_sizes, int n_in,
                              void* d_out, int out_size, void* d_ws, size_t ws_size,
                              hipStream_t stream) {
  const int*   prompts   = (const int*)d_in[0];
  const int*   p_len     = (const int*)d_in[1];
  const int*   ref_seq   = (const int*)d_in[2];
  const int*   r_len     = (const int*)d_in[3];
  const int*   text_seq  = (const int*)d_in[4];
  const int*   t_len     = (const int*)d_in[5];
  const float* ref_bert  = (const float*)d_in[6];
  const float* text_bert = (const float*)d_in[7];
  const float* text_emb  = (const float*)d_in[8];
  const float* audio_emb = (const float*)d_in[9];
  const float* bpw       = (const float*)d_in[10];
  const float* bpb       = (const float*)d_in[11];
  const float* alpha_t   = (const float*)d_in[12];
  const float* alpha_a   = (const float*)d_in[13];
  const float* qkv_w     = (const float*)d_in[14];
  const float* qkv_b     = (const float*)d_in[15];
  const float* out_w     = (const float*)d_in[16];
  const float* out_b     = (const float*)d_in[17];
  const float* ln1_w     = (const float*)d_in[18];
  const float* ln1_b     = (const float*)d_in[19];
  const float* ff1_w     = (const float*)d_in[20];
  const float* ff1_b     = (const float*)d_in[21];
  const float* ff2_w     = (const float*)d_in[22];
  const float* ff2_b     = (const float*)d_in[23];
  const float* ln2_w     = (const float*)d_in[24];
  const float* ln2_b     = (const float*)d_in[25];
  const float* pred_w    = (const float*)d_in[26];
  const float* pred_b    = (const float*)d_in[27];

  float* out = (float*)d_out;
  float* ws  = (float*)d_ws;

  float* h      = ws;                                   // 1024*512
  float* qkv    = h      + (size_t)TOTAL * DMODEL;      // 1024*1536
  float* attn_o = qkv    + (size_t)TOTAL * 3 * DMODEL;  // 1024*512
  float* tmp    = attn_o + (size_t)TOTAL * DMODEL;      // 1024*512
  float* ffbuf  = tmp    + (size_t)TOTAL * DMODEL;      // 1024*2048
  float* bertA  = ffbuf  + (size_t)TOTAL * FFDIM;       // 512*1024
  float* bproj  = bertA  + (size_t)XCAP * BERTD;        // 512*512
  f16_t* A16    = (f16_t*)(bproj + (size_t)XCAP * DMODEL);  // up to 1024*2048 halves
  f16_t* B16    = A16 + (size_t)TOTAL * FFDIM;              // up to 2048*512 halves

  float* kc = out + VOCAB + 4;                          // k_cache base
  float* vc = kc + (long long)NLAYER * CACHE_T * DMODEL;

  // --- cache pad zeroing (output contract: pad rows must be exactly 0) ---
  {
    long long per = (long long)NLAYER * MAXDEC * DMODEL;
    int blocks = (int)((per + 255) / 256);
    zero_pad_kernel<<<blocks, 256, 0, stream>>>(kc, vc);
  }

  // --- embedding path ---
  bertA_kernel<<<(XCAP * BERTD + 255) / 256, 256, 0, stream>>>(
      ref_bert, text_bert, r_len, t_len, bertA);
  cvt_kernel<<<(XCAP * BERTD / 4 + 255) / 256, 256, 0, stream>>>(
      bertA, A16, XCAP * BERTD);
  cvtT_kernel<<<dim3(DMODEL / 64, BERTD / 64), 256, 0, stream>>>(
      bpw, B16, BERTD, DMODEL);
  gemm_async<false, false, false><<<dim3(DMODEL / 128, XCAP / 128), 256, 0, stream>>>(
      A16, B16, nullptr, nullptr, bproj, XCAP, DMODEL, BERTD);
  embed_kernel<<<TOTAL, 256, 0, stream>>>(
      prompts, p_len, ref_seq, r_len, text_seq, t_len,
      text_emb, audio_emb, bproj, bpb, alpha_t, alpha_a, h);

  // --- transformer layers ---
  for (int l = 0; l < NLAYER; ++l) {
    const float* qw  = qkv_w + (size_t)l * DMODEL * 3 * DMODEL;
    const float* qb  = qkv_b + (size_t)l * 3 * DMODEL;
    const float* ow  = out_w + (size_t)l * DMODEL * DMODEL;
    const float* ob  = out_b + (size_t)l * DMODEL;
    const float* w1  = ff1_w + (size_t)l * DMODEL * FFDIM;
    const float* b1  = ff1_b + (size_t)l * FFDIM;
    const float* w2  = ff2_w + (size_t)l * FFDIM * DMODEL;
    const float* b2  = ff2_b + (size_t)l * DMODEL;
    const float* l1w = ln1_w + (size_t)l * DMODEL;
    const float* l1b = ln1_b + (size_t)l * DMODEL;
    const float* l2w = ln2_w + (size_t)l * DMODEL;
    const float* l2b = ln2_b + (size_t)l * DMODEL;
    float* kcl = kc + (long long)l * CACHE_T * DMODEL;
    float* vcl = vc + (long long)l * CACHE_T * DMODEL;

    // qkv = h @ qw + qb
    cvt_kernel<<<(TOTAL * DMODEL / 4 + 255) / 256, 256, 0, stream>>>(
        h, A16, TOTAL * DMODEL);
    cvtT_kernel<<<dim3(3 * DMODEL / 64, DMODEL / 64), 256, 0, stream>>>(
        qw, B16, DMODEL, 3 * DMODEL);
    gemm_async<true, false, false><<<dim3(3 * DMODEL / 128, TOTAL / 128), 256, 0, stream>>>(
        A16, B16, qb, nullptr, qkv, TOTAL, 3 * DMODEL, DMODEL);
    // emit k/v cache rows
    copy_kv_kernel<<<(TOTAL * DMODEL + 255) / 256, 256, 0, stream>>>(qkv, kcl, vcl);
    // attention
    attn_kernel<<<dim3(NHEAD, TOTAL / 64), 128, 0, stream>>>(
        qkv, attn_o, r_len, t_len, p_len);
    // tmp = h + attn_o @ ow + ob ; h = LN1(tmp)
    cvt_kernel<<<(TOTAL * DMODEL / 4 + 255) / 256, 256, 0, stream>>>(
        attn_o, A16, TOTAL * DMODEL);
    cvtT_kernel<<<dim3(DMODEL / 64, DMODEL / 64), 256, 0, stream>>>(
        ow, B16, DMODEL, DMODEL);
    gemm_async<true, false, true><<<dim3(DMODEL / 128, TOTAL / 128), 256, 0, stream>>>(
        A16, B16, ob, h, tmp, TOTAL, DMODEL, DMODEL);
    layernorm_kernel<<<TOTAL, 256, 0, stream>>>(tmp, l1w, l1b, h);
    // ff = relu(h @ w1 + b1) ; tmp = h + ff @ w2 + b2 ; h = LN2(tmp)
    cvt_kernel<<<(TOTAL * DMODEL / 4 + 255) / 256, 256, 0, stream>>>(
        h, A16, TOTAL * DMODEL);
    cvtT_kernel<<<dim3(FFDIM / 64, DMODEL / 64), 256, 0, stream>>>(
        w1, B16, DMODEL, FFDIM);
    gemm_async<true, true, false><<<dim3(FFDIM / 128, TOTAL / 128), 256, 0, stream>>>(
        A16, B16, b1, nullptr, ffbuf, TOTAL, FFDIM, DMODEL);
    cvt_kernel<<<(TOTAL * FFDIM / 4 + 255) / 256, 256, 0, stream>>>(
        ffbuf, A16, TOTAL * FFDIM);
    cvtT_kernel<<<dim3(DMODEL / 64, FFDIM / 64), 256, 0, stream>>>(
        w2, B16, FFDIM, DMODEL);
    gemm_async<true, false, true><<<dim3(DMODEL / 128, TOTAL / 128), 256, 0, stream>>>(
        A16, B16, b2, h, tmp, TOTAL, DMODEL, FFDIM);
    layernorm_kernel<<<TOTAL, 256, 0, stream>>>(tmp, l2w, l2b, h);
  }

  // --- logits / argmax / scalars ---
  predict_kernel<<<(VOCAB + 255) / 256, 256, 0, stream>>>(
      h, pred_w, pred_b, r_len, t_len, p_len, out);
  finalize_kernel<<<1, 256, 0, stream>>>(out, r_len, t_len, p_len, out);
}
